// GCN_1236950581656
// MI455X (gfx1250) — matrix operand order repaired and checked by reading, not compiled, not run on hardware
//
#include <hip/hip_runtime.h>

typedef __attribute__((ext_vector_type(16))) _Float16 v16h;
typedef __attribute__((ext_vector_type(8)))  _Float16 v8h;
typedef __attribute__((ext_vector_type(8)))  float    v8f;

#define GCN_N    69
#define GCN_NP   80          // padded node dim (M/N tiles of 16)
#define GCN_KP   96          // padded node dim as K (3 chunks of 32)
#define GCN_BT   (32 * 1024) // B*T
#define GCN_H1   32
#define GCN_H2   16

#define SLICES   8           // (b,t) slices per wave
#define WAVES    8           // waves per block
#define NBLK     (GCN_BT / (SLICES * WAVES))   // 512 blocks

// workspace layout in halves (contiguous, mirrored 1:1 into LDS):
//   [0,7680)     Ah  [80][96] row-major f16 normalized adjacency (symmetric), padded 0
//   [7680,8704)  W1t = W1^T [32 j][32 f] row-major, f cols 8..31 = 0
//   [8704,9216)  W2t = W2^T [16 h2][32 j] row-major
#define WS_A    0
#define WS_W1   7680
#define WS_W2   8704
#define WS_TOT  9216

// per-wave LDS (halves)
#define PW_X    1536   // sX: [16 f][96 m] = X^T row-major; reused as G^T [16 h2][96 m]
#define PW_Y    2560   // sY: [80 n][32 f] = Y1^T node-major; f cols 16..31 = 0 (one-time)
#define PW_H    2560   // sH: [80 n][32 j] = H^T node-major (fully rewritten each slice)
#define PW_TOT  (PW_X + PW_Y + PW_H)   // 6656 halves = 13312 B

// ---------------- prep: adj normalization + f16 weight packing ----------------
__global__ void gcn_prep(const float* __restrict__ adj,
                         const float* __restrict__ W1,
                         const float* __restrict__ W2,
                         _Float16* __restrict__ ws) {
    __shared__ float dis[GCN_N];
    const int tid = threadIdx.x;
    if (tid < GCN_N) {
        float s = 1.0f;  // self loop (A + I)
        for (int m = 0; m < GCN_N; ++m) s += adj[tid * GCN_N + m];
        dis[tid] = rsqrtf(s);
    }
    __syncthreads();
    for (int i = tid; i < GCN_NP * GCN_KP; i += (int)blockDim.x) {
        int r = i / GCN_KP, c = i % GCN_KP;
        float v = 0.0f;
        if (r < GCN_N && c < GCN_N) {
            float a = adj[r * GCN_N + c] + (r == c ? 1.0f : 0.0f);
            v = a * dis[r] * dis[c];
        }
        ws[WS_A + i] = (_Float16)v;
    }
    for (int i = tid; i < 32 * 32; i += (int)blockDim.x) {   // W1^T [j][f]
        int j = i / 32, f = i % 32;
        float v = (f < 8) ? W1[f * GCN_H1 + j] : 0.0f;
        ws[WS_W1 + i] = (_Float16)v;
    }
    for (int i = tid; i < 16 * 32; i += (int)blockDim.x) {   // W2^T [h2][j]
        int h2 = i / 32, j = i % 32;
        ws[WS_W2 + i] = (_Float16)W2[j * GCN_H2 + h2];
    }
}

// ---------------- WMMA helpers ----------------
__device__ __forceinline__ v8f wmma16(v16h a, v16h b, v8f c) {
    return __builtin_amdgcn_wmma_f32_16x16x32_f16(false, a, false, b, (short)0, c,
                                                  false, false);
}

// A fragment (16x32 f16) from row-major LDS: lane L: row=L%16, hi=L/16;
// half h -> K = kbase + hi*16 + h (one contiguous 32B run)
__device__ __forceinline__ v16h fragA(const _Float16* lds, int row0, int kbase,
                                      int ldk, int lane) {
    const int row = lane & 15, hi = lane >> 4;
    const _Float16* p = lds + (row0 + row) * ldk + kbase + hi * 16;
    v8h lo = *(const v8h*)p;
    v8h hv = *(const v8h*)(p + 8);
    v16h f;
#pragma unroll
    for (int i = 0; i < 8; ++i) { f[i] = lo[i]; f[i + 8] = hv[i]; }
    return f;
}

// B fragment (32x16 f16) from col-major LDS [col][K]: lane L: col=L%16, hi=L/16;
// half h -> K = kbase + hi*16 + h (one contiguous 32B run)
__device__ __forceinline__ v16h fragB(const _Float16* lds, int col0, int kbase,
                                      int ldk, int lane) {
    const int col = lane & 15, hi = lane >> 4;
    const _Float16* p = lds + (col0 + col) * ldk + kbase + hi * 16;
    v8h lo = *(const v8h*)p;
    v8h hv = *(const v8h*)(p + 8);
    v16h f;
#pragma unroll
    for (int i = 0; i < 8; ++i) { f[i] = lo[i]; f[i + 8] = hv[i]; }
    return f;
}

// pack 8 accumulator floats into 8 halves (optionally fused relu)
__device__ __forceinline__ v8h cvt8(v8f d, bool relu) {
    v8h o;
#pragma unroll
    for (int i = 0; i < 8; ++i) {
        float v = d[i];
        if (relu) v = v > 0.0f ? v : 0.0f;
        o[i] = (_Float16)v;
    }
    return o;
}

__device__ __forceinline__ float sigmoidf_(float v) {
    return 1.0f / (1.0f + __expf(-v));
}

// ---------------- fused GCN: transposed pipeline, 1 wave == SLICES slices ----
__global__ void __launch_bounds__(256)
gcn_main(const float* __restrict__ x, const _Float16* __restrict__ ws,
         float* __restrict__ out) {
    extern __shared__ _Float16 smem[];
    _Float16* sA  = smem;           // [80][96] row-major (symmetric adjacency)
    _Float16* sW1 = smem + WS_W1;   // W1^T [32][32]
    _Float16* sW2 = smem + WS_W2;   // W2^T [16][32]

    const int tid  = threadIdx.x;
    const int lane = tid & 31;
    const int wave = tid >> 5;
    const int col  = lane & 15;     // WMMA D col / B col
    const int hi   = lane >> 4;
    _Float16* wb = smem + WS_TOT + wave * PW_TOT;
    _Float16* sX = wb;              // X^T [16 f][96 m]; later G^T [16 h2][96 m]
    _Float16* sY = wb + PW_X;       // Y1^T [80 n][32 f]
    _Float16* sH = wb + PW_X + PW_Y;// H^T  [80 n][32 j]

    // cooperative staging of constants (LDS layout == ws layout), b128
    for (int i = tid; i < WS_TOT / 8; i += 256)
        ((v8h*)smem)[i] = ((const v8h*)ws)[i];

    v8h zh;
#pragma unroll
    for (int i = 0; i < 8; ++i) zh[i] = (_Float16)0.0f;
    // ONE-TIME zeros only. LDS starts undefined (could hold NaN patterns and
    // 0*NaN = NaN under WMMA), but once every value written is finite, stale
    // data is harmless: node cols >= 69 hit zero rows of bA, and f rows 8..15
    // propagate into Y1^T rows that W1^T's zeroed f>=8 columns annihilate.
#pragma unroll
    for (int j = 0; j < 6; ++j) ((v8h*)sX)[j * 32 + lane] = zh;
#pragma unroll
    for (int j = 0; j < 10; ++j) ((v8h*)sY)[j * 32 + lane] = zh;

    __syncthreads();

    // ---- register-resident invariant fragments ----
    // adjacency B-form frags: B[k, col] = A[n_out = col0+col][k] == A[k][n_out] (symmetric)
    v16h bA[15];
#pragma unroll
    for (int kc = 0; kc < 3; ++kc)
#pragma unroll
        for (int nt = 0; nt < 5; ++nt)
            bA[kc * 5 + nt] = fragB(sA, nt * 16, kc * 32, GCN_KP, lane);
    v16h aW1[2];
    aW1[0] = fragA(sW1, 0, 0, 32, lane);
    aW1[1] = fragA(sW1, 16, 0, 32, lane);
    v16h aW2 = fragA(sW2, 0, 0, 32, lane);

    v8f zf;
#pragma unroll
    for (int i = 0; i < 8; ++i) zf[i] = 0.0f;

    // strength-reduced per-slice base pointers
    const int bt0 = blockIdx.x * (WAVES * SLICES) + wave;
    const float* xb = x + (size_t)bt0 * 8;                    // + n*(BT*8) per node
    float* op = out + (size_t)bt0 * (GCN_N * GCN_H2);

    for (int s = 0; s < SLICES; ++s) {
        // stage X^T: sX[f][m] = x[m, bt, f]; 8 features of a node = one 32B run.
        // 69 nodes = 2 full lane rounds + 5-lane tail (branchless main body).
        {
            auto stage = [&](int n) {
                const float4* xp = reinterpret_cast<const float4*>(
                    xb + (size_t)n * (GCN_BT * 8));
                float4 a0 = xp[0], a1 = xp[1];
                sX[0 * GCN_KP + n] = (_Float16)a0.x;
                sX[1 * GCN_KP + n] = (_Float16)a0.y;
                sX[2 * GCN_KP + n] = (_Float16)a0.z;
                sX[3 * GCN_KP + n] = (_Float16)a0.w;
                sX[4 * GCN_KP + n] = (_Float16)a1.x;
                sX[5 * GCN_KP + n] = (_Float16)a1.y;
                sX[6 * GCN_KP + n] = (_Float16)a1.z;
                sX[7 * GCN_KP + n] = (_Float16)a1.w;
            };
            stage(lane);
            stage(lane + 32);
            if (lane < GCN_N - 64) stage(lane + 64);
        }

        // ---- Step 1: Y1^T = X^T @ A   (M=f 16, N=node 80, K=node 96): 15 WMMA
        v8f acc[5];
#pragma unroll
        for (int nt = 0; nt < 5; ++nt) acc[nt] = zf;
#pragma unroll
        for (int kc = 0; kc < 3; ++kc) {
            v16h aX = fragA(sX, 0, kc * 32, GCN_KP, lane);
#pragma unroll
            for (int nt = 0; nt < 5; ++nt)
                acc[nt] = wmma16(aX, bA[kc * 5 + nt], acc[nt]);
        }
        // packed b128 stores: sY[node][f], rows f = hi*8 + r contiguous
#pragma unroll
        for (int nt = 0; nt < 5; ++nt)
            *(v8h*)(sY + (nt * 16 + col) * 32 + hi * 8) = cvt8(acc[nt], false);

        // ---- Step 2: H^T = relu(W1^T @ Y1^T)  (M=j 32, N=node 80, K=f 32): 10 WMMA
#pragma unroll
        for (int nt = 0; nt < 5; ++nt) {
            v16h bY = fragB(sY, nt * 16, 0, 32, lane);
#pragma unroll
            for (int jt = 0; jt < 2; ++jt) {
                v8f h = wmma16(aW1[jt], bY, zf);
                *(v8h*)(sH + (nt * 16 + col) * 32 + jt * 16 + hi * 8) = cvt8(h, true);
            }
        }

        // ---- Step 3: G^T = W2^T @ H^T  (M=h2 16, N=node 80, K=j 32): 5 WMMA
#pragma unroll
        for (int nt = 0; nt < 5; ++nt) {
            v16h bH = fragB(sH, nt * 16, 0, 32, lane);
            v8f g = wmma16(aW2, bH, zf);
#pragma unroll
            for (int r = 0; r < 8; ++r)   // G^T row-major [h2][96 m] into sX
                sX[(hi * 8 + r) * GCN_KP + nt * 16 + col] = (_Float16)g[r];
        }

        // ---- Step 4: OUT^T = G^T @ A  (M=h2 16, N=node 80, K=node 96): 15 WMMA
#pragma unroll
        for (int nt = 0; nt < 5; ++nt) acc[nt] = zf;
#pragma unroll
        for (int kc = 0; kc < 3; ++kc) {
            v16h aG = fragA(sX, 0, kc * 32, GCN_KP, lane);
#pragma unroll
            for (int nt = 0; nt < 5; ++nt)
                acc[nt] = wmma16(aG, bA[kc * 5 + nt], acc[nt]);
        }

        // ---- sigmoid + vectorized output: lane col -> node, rows -> h2.
        // Tiles 0..3 cover n <= 63 < 69: unconditional. Only tile 4 is masked.
#pragma unroll
        for (int nt = 0; nt < 5; ++nt) {
            float4 o0, o1;
            o0.x = sigmoidf_(acc[nt][0]);
            o0.y = sigmoidf_(acc[nt][1]);
            o0.z = sigmoidf_(acc[nt][2]);
            o0.w = sigmoidf_(acc[nt][3]);
            o1.x = sigmoidf_(acc[nt][4]);
            o1.y = sigmoidf_(acc[nt][5]);
            o1.z = sigmoidf_(acc[nt][6]);
            o1.w = sigmoidf_(acc[nt][7]);
            float* p = op + (nt * 16 + col) * GCN_H2 + hi * 8;
            if (nt < 4) {
                *(float4*)p = o0;
                *(float4*)(p + 4) = o1;
            } else if (col < GCN_N - 64) {
                *(float4*)p = o0;
                *(float4*)(p + 4) = o1;
            }
        }

        xb += WAVES * 8;                         // next slice of this wave
        op += (size_t)WAVES * (GCN_N * GCN_H2);
    }
}

// ---------------- launch ----------------
extern "C" void kernel_launch(void* const* d_in, const int* in_sizes, int n_in,
                              void* d_out, int out_size, void* d_ws, size_t ws_size,
                              hipStream_t stream) {
    const float* x   = (const float*)d_in[0];  // [69, 32*1024, 8]
    const float* adj = (const float*)d_in[1];  // [69, 69]
    const float* W1  = (const float*)d_in[2];  // [8, 32]
    const float* W2  = (const float*)d_in[3];  // [32, 16]
    _Float16* ws = (_Float16*)d_ws;
    float* out = (float*)d_out;                // [32*1024, 69, 16]

    gcn_prep<<<1, 128, 0, stream>>>(adj, W1, W2, ws);

    const size_t lds_bytes = (size_t)(WS_TOT + WAVES * PW_TOT) * sizeof(_Float16); // 124928
    hipFuncSetAttribute((const void*)gcn_main,
                        hipFuncAttributeMaxDynamicSharedMemorySize,
                        (int)lds_bytes);
    // 512 blocks * 8 waves * 8 slices == 32768 (b,t) slices
    gcn_main<<<NBLK, 256, lds_bytes, stream>>>(x, ws, out);
}
